// PerceiverAttention_66305705116356
// MI455X (gfx1250) — matrix operand order repaired and checked by reading, not compiled, not run on hardware
//
#include <hip/hip_runtime.h>
#include <hip/hip_bf16.h>
#include <math.h>

typedef _Float16 half_t;
typedef __attribute__((ext_vector_type(16))) _Float16 v16h;
typedef __attribute__((ext_vector_type(8)))  _Float16 v8h;
typedef __attribute__((ext_vector_type(8)))  float    v8f;
typedef __attribute__((ext_vector_type(4)))  int      v4i;

#define BB    8
#define NQ    256
#define SS    4096
#define DIMM  1024
#define HH    16
#define DH    64
#define INNER 1024

// ---- CDNA5 async-to-LDS support (guarded: falls back to direct loads) -----
#if defined(__has_builtin)
#  if __has_builtin(__builtin_amdgcn_global_load_async_to_lds_b128)
#    define HAVE_ASYNC_LDS 1
#  endif
#endif
#ifndef HAVE_ASYNC_LDS
#  define HAVE_ASYNC_LDS 0
#endif

#if HAVE_ASYNC_LDS
// builtin signature (from hipcc diagnostic): (v4i addrspace(1)*, v4i addrspace(3)*, imm, imm)
typedef __attribute__((address_space(1))) v4i gv4i;
typedef __attribute__((address_space(3))) v4i lv4i;
#  if __has_builtin(__builtin_amdgcn_s_wait_asynccnt)
#    define WAIT_ASYNC(n)                         \
      do {                                        \
        __builtin_amdgcn_s_wait_asynccnt(n);      \
        asm volatile("" ::: "memory");            \
      } while (0)
#  else
#    define WAIT_ASYNC(n) asm volatile("s_wait_asynccnt %0" ::"i"(n) : "memory")
#  endif
#endif

// ---------------------------------------------------------------------------
// WMMA fragment helpers (this code's contiguous k order, k = 16*(lane/16) + s; wave32)
// ---------------------------------------------------------------------------
__device__ __forceinline__ v16h cat16(v8h lo, v8h hi) {
  union { v16h v; v8h h[2]; } u; u.h[0] = lo; u.h[1] = hi; return u.v;
}

// A-fragment 16x32 f16 from row-major [M x ld]: lane l<16 -> M=l, K=0..15;
// lane l>=16 -> M=l-16, K=16..31.  base points at (m0, k0).
__device__ __forceinline__ v16h load_a16(const half_t* base, int ld, int lane) {
  int l = lane & 15, kb = (lane >> 4) * 16;
  const half_t* p = base + (size_t)l * (size_t)ld + kb;
  v8h lo = *(const v8h*)p;
  v8h hi = *(const v8h*)(p + 8);
  return cat16(lo, hi);
}

// B-fragment 32x16 f16 from transposed (N-major, K-contiguous) storage [N x ld]:
// lane l<16 -> N=l, K=0..15; lane l>=16 -> N=l-16, K=16..31.  baseT points at (n0, k0).
__device__ __forceinline__ v16h load_b16(const half_t* baseT, size_t ld, int lane) {
  int l = lane & 15, kb = (lane >> 4) * 16;
  return *(const v16h*)(baseT + (size_t)l * ld + kb);
}

__device__ __forceinline__ v8f wmma_f16(v16h a, v16h b, v8f c) {
  return __builtin_amdgcn_wmma_f32_16x16x32_f16(false, a, false, b, (short)0, c,
                                                false, false);
}

// ---------------------------------------------------------------------------
// Kernel: convert f32 weight [K x N] -> f16 transposed [N x K]
// ---------------------------------------------------------------------------
__global__ __launch_bounds__(256)
void wcvt_kernel(const float* __restrict__ W, half_t* __restrict__ Wt,
                 int K, int N) {
  size_t total = (size_t)K * (size_t)N;
  for (size_t idx = (size_t)blockIdx.x * blockDim.x + threadIdx.x; idx < total;
       idx += (size_t)gridDim.x * blockDim.x) {
    size_t k = idx / (size_t)N, n = idx % (size_t)N;
    Wt[n * (size_t)K + k] = (half_t)W[idx];
  }
}

// ---------------------------------------------------------------------------
// Kernel: LayerNorm one row of 1024 per block, f32 in -> f16 out
// ---------------------------------------------------------------------------
__global__ __launch_bounds__(256)
void ln_kernel(const float* __restrict__ X, const float* __restrict__ g,
               const float* __restrict__ bta, half_t* __restrict__ Y) {
  __shared__ float ssum[256], ssq[256];
  int t = threadIdx.x;
  const float* x = X + (size_t)blockIdx.x * DIMM;
  float v0 = x[t], v1 = x[t + 256], v2 = x[t + 512], v3 = x[t + 768];
  float s = v0 + v1 + v2 + v3;
  float q = v0 * v0 + v1 * v1 + v2 * v2 + v3 * v3;
  ssum[t] = s; ssq[t] = q;
  __syncthreads();
  for (int st = 128; st > 0; st >>= 1) {
    if (t < st) { ssum[t] += ssum[t + st]; ssq[t] += ssq[t + st]; }
    __syncthreads();
  }
  float mu = ssum[0] * (1.0f / DIMM);
  float var = ssq[0] * (1.0f / DIMM) - mu * mu;
  float rstd = rsqrtf(var + 1e-5f);
  half_t* y = Y + (size_t)blockIdx.x * DIMM;
  y[t]       = (half_t)((v0 - mu) * rstd * g[t]       + bta[t]);
  y[t + 256] = (half_t)((v1 - mu) * rstd * g[t + 256] + bta[t + 256]);
  y[t + 512] = (half_t)((v2 - mu) * rstd * g[t + 512] + bta[t + 512]);
  y[t + 768] = (half_t)((v3 - mu) * rstd * g[t + 768] + bta[t + 768]);
}

// ---------------------------------------------------------------------------
// Kernel: WMMA GEMM  C[MxN] = A[MxK](f16,row-major) * Bt[NxK](f16)^T
// Block = 256 threads = 8 waves; block tile 64(M) x 128(N); wave tile 16x64.
// C16 != nullptr -> f16 output; else f32 output + bias.
// ---------------------------------------------------------------------------
__global__ __launch_bounds__(256)
void gemm_wmma(const half_t* __restrict__ A, const half_t* __restrict__ Bt,
               half_t* __restrict__ C16, float* __restrict__ C32,
               const float* __restrict__ bias, int M, int N, int K) {
  int lane = threadIdx.x & 31;
  int w    = threadIdx.x >> 5;            // 0..7
  int m0 = blockIdx.y * 64  + (w & 3) * 16;
  int n0 = blockIdx.x * 128 + (w >> 2) * 64;
  int l  = lane & 15;

  v8f acc[4] = {};
  const half_t* arow = A + (size_t)m0 * (size_t)K;
  for (int k0 = 0; k0 < K; k0 += 32) {
    v16h a = load_a16(arow + k0, K, lane);
#pragma unroll
    for (int tt = 0; tt < 4; ++tt) {
      v16h b = load_b16(Bt + (size_t)(n0 + 16 * tt) * (size_t)K + k0,
                        (size_t)K, lane);
      acc[tt] = wmma_f16(a, b, acc[tt]);
    }
  }

  int mb = (lane >> 4) * 8;
#pragma unroll
  for (int tt = 0; tt < 4; ++tt) {
    int col = n0 + 16 * tt + l;
    if (C16) {
#pragma unroll
      for (int r = 0; r < 8; ++r)
        C16[(size_t)(m0 + mb + r) * (size_t)N + col] = (half_t)acc[tt][r];
    } else {
      float bv = bias ? bias[col] : 0.0f;
#pragma unroll
      for (int r = 0; r < 8; ++r)
        C32[(size_t)(m0 + mb + r) * (size_t)N + col] = acc[tt][r] + bv;
    }
  }
}

// ---------------------------------------------------------------------------
// Kernel: RoPE (in-place on K half of kv) + transpose V -> Vt[bh*64+d][S]
// One block per (b,s) row of kv [B*S x 2048].
// ---------------------------------------------------------------------------
__global__ __launch_bounds__(256)
void rope_vt_kernel(half_t* __restrict__ kv, const float* __restrict__ cosb,
                    const float* __restrict__ sinb, half_t* __restrict__ Vt) {
  int row = blockIdx.x;                 // b*S + s
  int b = row >> 12, s = row & (SS - 1);
  half_t* kvrow = kv + (size_t)row * 2048;
  int t = threadIdx.x;
  // RoPE on K half: 512 (h,d) pairs, d in [0,32)
  for (int i = t; i < 512; i += 256) {
    int h = i >> 5, d = i & 31;
    float c  = cosb[s * 32 + d];
    float sn = sinb[s * 32 + d];
    float x1 = (float)kvrow[h * 64 + d];
    float x2 = (float)kvrow[h * 64 + d + 32];
    kvrow[h * 64 + d]      = (half_t)(x1 * c - x2 * sn);
    kvrow[h * 64 + d + 32] = (half_t)(x2 * c + x1 * sn);
  }
  // V transpose: Vt[(b*16+h)*64 + d][s] = V[b,s,h,d]
  for (int i = t; i < 1024; i += 256) {
    int h = i >> 6, d = i & 63;
    Vt[((size_t)((b * HH + h) * 64 + d)) * SS + s] = kvrow[1024 + i];
  }
}

// ---------------------------------------------------------------------------
// Kernel: streaming attention.  2 blocks per (b,h); 8 waves each, one 16-row
// Q tile per wave.  Chunks of 32 keys: K tile staged into LDS with
// double-buffered GLOBAL_LOAD_ASYNC_TO_LDS_B128 (ASYNCcnt split-barrier),
// 4 WMMA (QK^T) + exp + LDS layout swap + 4 WMMA (P*V).
// Clip to +-11 makes exp bounded -> fixed-base online softmax, no running max.
// ---------------------------------------------------------------------------
__global__ __launch_bounds__(256)
void attn_kernel(const half_t* __restrict__ Q, const half_t* __restrict__ kv,
                 const half_t* __restrict__ Vt, half_t* __restrict__ O) {
  __shared__ __align__(32) half_t Pt[8][16 * 32];       // per-wave P tile (8 KB)
#if HAVE_ASYNC_LDS
  __shared__ __align__(32) half_t Kt[8][2][32 * 64];    // per-wave dbl K tile (64 KB)
#endif
  int bh = blockIdx.x >> 1;
  int b = bh >> 4, h = bh & 15;
  int lane = threadIdx.x & 31;
  int w    = threadIdx.x >> 5;               // 0..7
  int qt   = (blockIdx.x & 1) * 8 + w;       // 0..15: q-tile index
  int l    = lane & 15;
  int mb   = (lane >> 4) * 8;

  const half_t* Qbase  = Q  + (size_t)(b * NQ + qt * 16) * INNER + h * 64;
  const half_t* Kbase  = kv + (size_t)b * SS * 2048 + h * 64;      // row stride 2048
  const half_t* Vtbase = Vt + (size_t)bh * 64 * SS;                // row stride SS
  half_t* myP = &Pt[w][0];

  v16h aq0 = load_a16(Qbase,      INNER, lane);   // K 0..31 of d
  v16h aq1 = load_a16(Qbase + 32, INNER, lane);   // K 32..63 of d

  v8f acc[4] = {};
  float rs = 0.0f;                                // per-lane partial row-sum

#if HAVE_ASYNC_LDS
  half_t* kt[2] = { &Kt[w][0][0], &Kt[w][1][0] };
  // stage one 32x64 f16 K tile (4 KB) = 256 x 16B segments, 8 per lane
  auto stageK = [&](int s0, half_t* dst) {
#pragma unroll
    for (int it = 0; it < 8; ++it) {
      int seg = it * 32 + lane;
      int row = seg >> 3;            // 0..31
      int so  = (seg & 7) * 8;       // halves (16B granules) within row
      __builtin_amdgcn_global_load_async_to_lds_b128(
          (gv4i*)(Kbase + (size_t)(s0 + row) * 2048 + so),
          (lv4i*)(dst + row * 64 + so), 0, 0);
    }
  };
  stageK(0, kt[0]);
#endif

  for (int s0 = 0; s0 < SS; s0 += 32) {
#if HAVE_ASYNC_LDS
    half_t* cur = kt[(s0 >> 5) & 1];
    half_t* nxt = kt[((s0 >> 5) & 1) ^ 1];
    if (s0 + 32 < SS) { stageK(s0 + 32, nxt); WAIT_ASYNC(8); }
    else              { WAIT_ASYNC(0); }
#else
    if (s0 + 32 < SS)  // pull next K tile toward the WGP while we compute
      __builtin_prefetch(Kbase + (size_t)(s0 + 32 + l) * 2048, 0, 0);
#endif
#pragma unroll
    for (int tt = 0; tt < 2; ++tt) {
#if HAVE_ASYNC_LDS
      const half_t* kp = cur + (size_t)(16 * tt) * 64;
      v16h bk0 = load_b16(kp,      64, lane);     // d 0..31
      v16h bk1 = load_b16(kp + 32, 64, lane);     // d 32..63
#else
      const half_t* kp = Kbase + (size_t)(s0 + 16 * tt) * 2048;
      v16h bk0 = load_b16(kp,      2048, lane);   // d 0..31
      v16h bk1 = load_b16(kp + 32, 2048, lane);   // d 32..63
#endif
      v8f c = {};
      c = wmma_f16(aq0, bk0, c);
      c = wmma_f16(aq1, bk1, c);
      int col = l + 16 * tt;
#pragma unroll
      for (int r = 0; r < 8; ++r) {
        float sc = c[r] * 0.125f;                 // DHEAD^-0.5
        sc = fminf(fmaxf(sc, -11.0f), 11.0f);     // reference clip
        myP[(mb + r) * 32 + col] = (half_t)__expf(sc);  // in [1.7e-5, 6e4]: fits f16
      }
    }
    // intra-wave LDS write->read ordering (DS ops are in-order per wave)
    asm volatile("s_wait_dscnt 0" ::: "memory");
    v16h pa = load_a16(myP, 32, lane);            // P as A-fragment
#pragma unroll
    for (int i = 0; i < 16; ++i) rs += (float)pa[i];
#pragma unroll
    for (int tt = 0; tt < 4; ++tt) {
      v16h bv = load_b16(Vtbase + (size_t)(16 * tt) * SS + s0, (size_t)SS, lane);
      acc[tt] = wmma_f16(pa, bv, acc[tt]);
    }
  }

  // row-sum: lane m (and m+16) hold partials for local row m
  float rtot = rs + __shfl_xor(rs, 16);
  half_t* Obase = O + (size_t)(b * NQ + qt * 16) * INNER + h * 64;
#pragma unroll
  for (int r = 0; r < 8; ++r) {
    float inv = 1.0f / __shfl(rtot, mb + r);
#pragma unroll
    for (int tt = 0; tt < 4; ++tt)
      Obase[(size_t)(mb + r) * INNER + 16 * tt + l] = (half_t)(acc[tt][r] * inv);
  }
}

// ---------------------------------------------------------------------------
// Host orchestration
// ---------------------------------------------------------------------------
extern "C" void kernel_launch(void* const* d_in, const int* in_sizes, int n_in,
                              void* d_out, int out_size, void* d_ws, size_t ws_size,
                              hipStream_t stream) {
  const float* latents = (const float*)d_in[0];
  const float* context = (const float*)d_in[1];
  const float* cosb    = (const float*)d_in[2];
  const float* sinb    = (const float*)d_in[3];
  const float* ln_lat_g = (const float*)d_in[4];
  const float* ln_lat_b = (const float*)d_in[5];
  const float* ln_ctx_g = (const float*)d_in[6];
  const float* ln_ctx_b = (const float*)d_in[7];
  const float* Wq  = (const float*)d_in[8];
  const float* Wkv = (const float*)d_in[9];
  const float* Wo  = (const float*)d_in[10];
  const float* bo  = (const float*)d_in[11];
  float* out = (float*)d_out;

  char* ws = (char*)d_ws;
  size_t off = 0;
  auto carve = [&](size_t bytes) -> char* {
    char* p = ws + off;
    off += (bytes + 255) & ~(size_t)255;
    return p;
  };
  half_t* WqT    = (half_t*)carve((size_t)2 * DIMM * INNER);        //  2 MB
  half_t* WkvT   = (half_t*)carve((size_t)2 * DIMM * 2 * INNER);    //  4 MB
  half_t* WoT    = (half_t*)carve((size_t)2 * INNER * DIMM);        //  2 MB
  half_t* lat16  = (half_t*)carve((size_t)2 * BB * NQ * DIMM);      //  4 MB
  half_t* ctx16  = (half_t*)carve((size_t)2 * BB * SS * DIMM);      // 64 MB
  half_t* q16    = (half_t*)carve((size_t)2 * BB * NQ * INNER);     //  4 MB
  half_t* kv16   = (half_t*)carve((size_t)2 * BB * SS * 2 * INNER); // 128 MB
  half_t* Vt     = (half_t*)carve((size_t)2 * BB * HH * DH * SS);   // 64 MB
  half_t* attn16 = (half_t*)carve((size_t)2 * BB * NQ * INNER);     //  4 MB

  // 1. weights -> f16, transposed (N-major, K-contiguous)
  wcvt_kernel<<<2048, 256, 0, stream>>>(Wq,  WqT,  DIMM, INNER);
  wcvt_kernel<<<2048, 256, 0, stream>>>(Wkv, WkvT, DIMM, 2 * INNER);
  wcvt_kernel<<<2048, 256, 0, stream>>>(Wo,  WoT,  INNER, DIMM);

  // 2. layernorms -> f16 activations
  ln_kernel<<<BB * NQ, 256, 0, stream>>>(latents, ln_lat_g, ln_lat_b, lat16);
  ln_kernel<<<BB * SS, 256, 0, stream>>>(context, ln_ctx_g, ln_ctx_b, ctx16);

  // 3. Q = lat * Wq   (2048 x 1024 x 1024)
  gemm_wmma<<<dim3(INNER / 128, BB * NQ / 64), 256, 0, stream>>>(
      lat16, WqT, q16, nullptr, nullptr, BB * NQ, INNER, DIMM);

  // 4. KV = ctx * Wkv (32768 x 2048 x 1024)
  gemm_wmma<<<dim3(2 * INNER / 128, BB * SS / 64), 256, 0, stream>>>(
      ctx16, WkvT, kv16, nullptr, nullptr, BB * SS, 2 * INNER, DIMM);

  // 5. RoPE on K (in place) + transpose V
  rope_vt_kernel<<<BB * SS, 256, 0, stream>>>(kv16, cosb, sinb, Vt);

  // 6. attention: 2 blocks per (b,h), 8 waves each
  attn_kernel<<<BB * HH * 2, 256, 0, stream>>>(q16, kv16, Vt, attn16);

  // 7. out = attn * Wo + bo  (2048 x 1024 x 1024), f32 output
  gemm_wmma<<<dim3(DIMM / 128, BB * NQ / 64), 256, 0, stream>>>(
      attn16, WoT, nullptr, out, bo, BB * NQ, DIMM, INNER);

  (void)in_sizes; (void)n_in; (void)out_size; (void)ws_size;
}
